// AccRNNCell_38628935860387
// MI455X (gfx1250) — matrix operand order repaired and checked by reading, not compiled, not run on hardware
//
#include <hip/hip_runtime.h>

// ---------------------------------------------------------------------------
// AccRNN on MI455X (gfx1250): persistent per-batch-tile workgroups, bf16 WMMA
// (V_WMMA_F32_16X16X32_BF16) with fp32 accumulation, state resident in LDS.
// ---------------------------------------------------------------------------

typedef __attribute__((ext_vector_type(16))) __bf16       v16bf;
typedef __attribute__((ext_vector_type(8)))  float        v8f;
typedef __attribute__((ext_vector_type(4)))  unsigned int v4u;

union Frag { v4u q[2]; v16bf v; };

static constexpr int cL = 3;
static constexpr int cU = 512;
static constexpr int cP = 32;
static constexpr int cF = 64;
static constexpr int cB = 512;
static constexpr int cT = 512;

// packed bf16 weight offsets inside workspace (ushort units)
static constexpr int oWA   = 0;                         // 3 * 512*512
static constexpr int oWB0  = 3 * cU * cU;               // 96*512
static constexpr int oWBr  = oWB0 + (cF + cP) * cU;     // 2 * 512*512
static constexpr int oWC   = oWBr + 2 * cU * cU;        // 3 * 512*512
static constexpr int oWout = oWC + 3 * cU * cU;         // 512*32

__device__ __forceinline__ unsigned short f2bf(float f) {
  unsigned int b = __float_as_uint(f);
  return (unsigned short)((b + 0x7FFFu + ((b >> 16) & 1u)) >> 16);  // RNE
}

// A fragment (16x32 bf16, M rows = batch tile) from LDS row-major [16][W=512].
// Lane l: row = l&15; K pairs: v0..3 -> 8*hi + {0..7}, v4..7 -> 16 + 8*hi + {0..7}
__device__ __forceinline__ v16bf load_a_lds(const unsigned short* s, int rowoff,
                                            int k0, int hi8) {
  Frag f;
  f.q[0] = *(const v4u*)(s + rowoff + k0 + hi8);
  f.q[1] = *(const v4u*)(s + rowoff + k0 + 16 + hi8);
  return f.v;
}

// B fragment (32x16 bf16), pre-swizzled in workspace: frag-major, 32 lanes * 16
// ushorts contiguous per lane -> two coalesced 128-bit global loads.
__device__ __forceinline__ v16bf load_b_glb(const unsigned short* w, int frag,
                                            int lane) {
  const v4u* p = (const v4u*)(w + (size_t)frag * 512 + lane * 16);
  Frag f;
  f.q[0] = p[0];
  f.q[1] = p[1];
  return f.v;
}

// fp32 [Kdim][Ndim] row-major -> bf16 WMMA B-fragment order.
__global__ void pack_w(const float* __restrict__ src,
                       unsigned short* __restrict__ dst, int Kdim, int Ndim) {
  int i = blockIdx.x * blockDim.x + threadIdx.x;
  if (i >= Kdim * Ndim) return;
  int k = i / Ndim, n = i - k * Ndim;
  int Ktiles = Kdim >> 5;
  int ntile = n >> 4, nin = n & 15, ktile = k >> 5, kin = k & 31;
  int lane = nin + (((kin >> 3) & 1) << 4);
  int kk = (kin & 7) + ((kin >> 4) << 3);
  int idx = (((ntile * Ktiles + ktile) * 32 + lane) * 8 + (kk >> 1)) * 2 + (kk & 1);
  dst[idx] = f2bf(src[i]);
}

__global__ __launch_bounds__(256, 1) void accrnn_main(
    const float* __restrict__ x, const unsigned short* __restrict__ pw,
    const float* __restrict__ bA, const float* __restrict__ bB0,
    const float* __restrict__ bBr, const float* __restrict__ bC,
    const float* __restrict__ bout, float* __restrict__ out) {
  __shared__ unsigned short sh_hs[cL * 16 * cU];  // 48 KB: recurrent states
  __shared__ unsigned short sh_prev[16 * cU];     // 16 KB: inter-layer / concat

  const int tid  = threadIdx.x;
  const int wave = tid >> 5;
  const int lane = tid & 31;
  const int nl   = lane & 15;
  const int hi8  = (lane >> 4) * 8;  // A-frag K sub-offset
  const int mo   = (lane >> 4) * 8;  // C-frag row offset
  const int b0   = blockIdx.x * 16;  // this WG's batch rows

  for (int i = tid; i < cL * 16 * cU; i += 256) sh_hs[i] = 0;

  float accreg[8];  // running prediction accumulator (valid in waves 0,1)
#pragma unroll
  for (int j = 0; j < 8; ++j) accreg[j] = 0.0f;

  __syncthreads();

  for (int t = 0; t < cT; ++t) {
    // concat(x_t, acc) -> prev[:, 0:96]  (bf16)
    for (int e = tid; e < 16 * cF; e += 256) {
      int r = e >> 6, c = e & 63;
      sh_prev[r * cU + c] =
          f2bf(x[(size_t)(b0 + r) * (cT * cF) + (size_t)t * cF + c]);
    }
    if (wave < 2) {
      int n = wave * 16 + nl;
#pragma unroll
      for (int e = 0; e < 8; ++e)
        sh_prev[(mo + e) * cU + cF + n] = f2bf(accreg[e]);
    }
    __syncthreads();

    for (int i = 0; i < cL; ++i) {
      unsigned short* hsi = sh_hs + i * 16 * cU;
      const unsigned short* pWA = pw + oWA + i * cU * cU;
      const unsigned short* pWB =
          (i == 0) ? (pw + oWB0) : (pw + oWBr + (i - 1) * cU * cU);
      const unsigned short* pWC = pw + oWC + i * cU * cU;
      const float* bAl = bA + i * cU;
      const float* bBl = (i == 0) ? bB0 : (bBr + (i - 1) * cU);
      const float* bCl = bC + i * cU;
      const int KtB = (i == 0) ? 3 : 16;  // 96-wide concat vs 512-wide

      v8f accv[4];
#pragma unroll
      for (int j = 0; j < 4; ++j) accv[j] = {};

      // latent_state: hs_i @ WA_i   (K = 512)
      for (int kt = 0; kt < 16; ++kt) {
        v16bf a = load_a_lds(hsi, nl * cU, kt * 32, hi8);
#pragma unroll
        for (int j = 0; j < 4; ++j) {
          v16bf b = load_b_glb(pWA, (wave * 4 + j) * 16 + kt, lane);
          accv[j] = __builtin_amdgcn_wmma_f32_16x16x32_bf16(
              false, a, false, b, (short)0, accv[j], false, false);
        }
      }
      // latent_input: prev @ WB  (accumulates into same C — they are summed)
      for (int kt = 0; kt < KtB; ++kt) {
        v16bf a = load_a_lds(sh_prev, nl * cU, kt * 32, hi8);
#pragma unroll
        for (int j = 0; j < 4; ++j) {
          v16bf b = load_b_glb(pWB, (wave * 4 + j) * KtB + kt, lane);
          accv[j] = __builtin_amdgcn_wmma_f32_16x16x32_bf16(
              false, a, false, b, (short)0, accv[j], false, false);
        }
      }
      __syncthreads();  // everyone done reading old hs_i / prev
#pragma unroll
      for (int j = 0; j < 4; ++j) {
        int n = (wave * 4 + j) * 16 + nl;
        float bias = bAl[n] + bBl[n];
#pragma unroll
        for (int e = 0; e < 8; ++e)
          hsi[(mo + e) * cU + n] = f2bf(accv[j][e] + bias);
      }
      __syncthreads();  // new state visible to all waves

      // layer output: new_state @ WC_i  (K = 512)
#pragma unroll
      for (int j = 0; j < 4; ++j) accv[j] = {};
      for (int kt = 0; kt < 16; ++kt) {
        v16bf a = load_a_lds(hsi, nl * cU, kt * 32, hi8);
#pragma unroll
        for (int j = 0; j < 4; ++j) {
          v16bf b = load_b_glb(pWC, (wave * 4 + j) * 16 + kt, lane);
          accv[j] = __builtin_amdgcn_wmma_f32_16x16x32_bf16(
              false, a, false, b, (short)0, accv[j], false, false);
        }
      }
      __syncthreads();  // all reads of old prev done
#pragma unroll
      for (int j = 0; j < 4; ++j) {
        int n = (wave * 4 + j) * 16 + nl;
        float bias = bCl[n];
#pragma unroll
        for (int e = 0; e < 8; ++e)
          sh_prev[(mo + e) * cU + n] = f2bf(accv[j][e] + bias);
      }
      __syncthreads();
    }

    // prediction head: prev @ Wout (512x32) on waves 0,1; acc += res
    if (wave < 2) {
      v8f o = {};
      for (int kt = 0; kt < 16; ++kt) {
        v16bf a = load_a_lds(sh_prev, nl * cU, kt * 32, hi8);
        v16bf b = load_b_glb(pw + oWout, wave * 16 + kt, lane);
        o = __builtin_amdgcn_wmma_f32_16x16x32_bf16(false, a, false, b, (short)0,
                                                    o, false, false);
      }
      int n = wave * 16 + nl;
      float bias = bout[n];
#pragma unroll
      for (int e = 0; e < 8; ++e) {
        float val = o[e] + bias;
        out[(size_t)(b0 + mo + e) * (cT * cP) + (size_t)t * cP + n] = val;
        accreg[e] += val;  // carry for next step's concat
      }
    }
    __syncthreads();  // prev reads done before next step overwrites it
  }
}

extern "C" void kernel_launch(void* const* d_in, const int* in_sizes, int n_in,
                              void* d_out, int out_size, void* d_ws,
                              size_t ws_size, hipStream_t stream) {
  const float* x    = (const float*)d_in[0];
  const float* WA   = (const float*)d_in[1];
  const float* bA   = (const float*)d_in[2];
  const float* WB0  = (const float*)d_in[3];
  const float* bB0  = (const float*)d_in[4];
  const float* WBr  = (const float*)d_in[5];
  const float* bBr  = (const float*)d_in[6];
  const float* WC   = (const float*)d_in[7];
  const float* bC   = (const float*)d_in[8];
  const float* Wout = (const float*)d_in[9];
  const float* bout = (const float*)d_in[10];
  float* out = (float*)d_out;
  unsigned short* ws = (unsigned short*)d_ws;

  auto nb = [](int n) { return (n + 255) / 256; };
  // fp32 -> bf16 + WMMA-B-fragment swizzle (re-done every call: deterministic,
  // graph-capture safe, ~4.3 MB total)
  for (int i = 0; i < 3; ++i)
    pack_w<<<nb(cU * cU), 256, 0, stream>>>(WA + i * cU * cU,
                                            ws + oWA + i * cU * cU, cU, cU);
  pack_w<<<nb((cF + cP) * cU), 256, 0, stream>>>(WB0, ws + oWB0, cF + cP, cU);
  for (int i = 0; i < 2; ++i)
    pack_w<<<nb(cU * cU), 256, 0, stream>>>(WBr + i * cU * cU,
                                            ws + oWBr + i * cU * cU, cU, cU);
  for (int i = 0; i < 3; ++i)
    pack_w<<<nb(cU * cU), 256, 0, stream>>>(WC + i * cU * cU,
                                            ws + oWC + i * cU * cU, cU, cU);
  pack_w<<<nb(cU * cP), 256, 0, stream>>>(Wout, ws + oWout, cU, cP);

  // persistent recurrent kernel: 32 workgroups x 16 batch rows, full T loop
  accrnn_main<<<cB / 16, 256, 0, stream>>>(x, ws, bA, bB0, bBr, bC, bout, out);
}
